// TGCN_38053410243242
// MI455X (gfx1250) — matrix operand order repaired and checked by reading, not compiled, not run on hardware
//
#include <hip/hip_runtime.h>
#include <hip/hip_bf16.h>

// TGCN on MI455X (gfx1250): heavy matmuls via v_wmma_f32_16x16x32_bf16,
// adjacency GEMM A-tiles staged with global_load_async_to_lds_b128 (ASYNCcnt)
// in a double-buffered LDS pipeline.
// B=32, S=12, N=2048, F=1, H=128, G=64, O=1, P=3 (hardcoded; shapes fixed).

typedef __bf16 bf16;
typedef __attribute__((ext_vector_type(8)))  __bf16 v8bf;
typedef __attribute__((ext_vector_type(16))) __bf16 v16bf;
typedef __attribute__((ext_vector_type(8)))  float  v8f;

__device__ __forceinline__ v16bf cat16(v8bf lo, v8bf hi) {
  return __builtin_shufflevector(lo, hi, 0,1,2,3,4,5,6,7,8,9,10,11,12,13,14,15);
}

__device__ __forceinline__ v8f wmma_bf16(v16bf a, v16bf b, v8f c) {
  // (neg_a, A, neg_b, B, c_mod, C, reuse_a, reuse_b)
  return __builtin_amdgcn_wmma_f32_16x16x32_bf16(false, a, false, b, (short)0, c,
                                                 false, false);
}

__device__ __forceinline__ float sigmoidf(float x) {
  return 1.0f / (1.0f + __expf(-x));
}

// Async DMA: 16 bytes per lane, global -> LDS, tracked by ASYNCcnt.
__device__ __forceinline__ void async_copy16(unsigned lds_addr, const void* gaddr) {
  asm volatile("global_load_async_to_lds_b128 %0, %1, off"
               :: "v"(lds_addr), "v"((unsigned long long)(uintptr_t)gaddr)
               : "memory");
}

// ---------------------------------------------------------------------------
// General bf16 GEMM: C(MxNc) = A(MxK) @ B(KxNc), row-major, f32 accumulate.
// MODE 0: store bf16. MODE 1: +bias[c&bmask], ReLU, bf16. MODE 2: +bias, bf16.
// Block: 256 thr = 8 waves; block tile 128x64; wave tile 16x64; K-step 32.
// A tile: async DMA to LDS, double-buffered. B tile: vector load + LDS transpose.
// ---------------------------------------------------------------------------
constexpr int BM = 128, BN = 64, BK = 32, LPAD = 40;

template <int MODE>
__global__ __launch_bounds__(256)
void wmma_gemm_bf16(const bf16* __restrict__ A, const bf16* __restrict__ Bm,
                    bf16* __restrict__ C, const float* __restrict__ bias,
                    int M, int Nc, int K, int bmask) {
  __shared__ bf16 sA[2][BM * LPAD];   // A tiles, row-major, padded rows
  __shared__ bf16 sB[2][BN * LPAD];   // B tiles transposed: sB[buf][col][k]

  const int m0   = blockIdx.y * BM;
  const int n0   = blockIdx.x * BN;
  const int t    = threadIdx.x;
  const int lane = t & 31;
  const int wid  = t >> 5;

  v8f acc[4] = {};

  // cooperative-load coordinates
  const int row  = t >> 2;         // 0..63 (and +64)
  const int kc   = (t & 3) * 8;    // A K-chunk within tile
  const int krow = t >> 3;         // 0..31 (B k row)
  const int cc   = (t & 7) * 8;    // B col chunk

  // fragment-gather coordinates (lane half h holds K 16h..16h+15)
  const int arow  = wid * 16 + (lane & 15);
  const int koff  = (lane >> 4) * 16;  // A frag: contiguous K chunk of 16
  const int bkoff = (lane >> 4) * 16;  // B frag: contiguous K chunk of 16

  auto issueA = [&](int buf, int k0) {
    async_copy16((unsigned)(uintptr_t)&sA[buf][row * LPAD + kc],
                 A + (size_t)(m0 + row) * K + k0 + kc);
    async_copy16((unsigned)(uintptr_t)&sA[buf][(row + 64) * LPAD + kc],
                 A + (size_t)(m0 + row + 64) * K + k0 + kc);
  };
  auto loadB = [&](int k0) -> v8bf {
    return *(const v8bf*)(Bm + (size_t)(k0 + krow) * Nc + n0 + cc);
  };

  // prologue: prefetch tile 0
  issueA(0, 0);
  v8bf breg = loadB(0);

  const int nk = K / BK;
  for (int ik = 0; ik < nk; ++ik) {
    const int buf = ik & 1;
    // stage B (transposed) into LDS
#pragma unroll
    for (int j = 0; j < 8; ++j) sB[buf][(cc + j) * LPAD + krow] = breg[j];

    const bool has_next = (ik + 1 < nk);
    if (has_next) {
      issueA(buf ^ 1, (ik + 1) * BK);     // DMA next A tile while we compute
      breg = loadB((ik + 1) * BK);        // next B tile into regs
      asm volatile("s_wait_asynccnt 0x2" ::: "memory");  // current tile done
    } else {
      asm volatile("s_wait_asynccnt 0x0" ::: "memory");
    }
    __syncthreads();

    v16bf a = cat16(*(const v8bf*)(&sA[buf][arow * LPAD + koff]),
                    *(const v8bf*)(&sA[buf][arow * LPAD + koff + 8]));
#pragma unroll
    for (int cb = 0; cb < 4; ++cb) {
      int bcol = cb * 16 + (lane & 15);
      v16bf b = cat16(*(const v8bf*)(&sB[buf][bcol * LPAD + bkoff]),
                      *(const v8bf*)(&sB[buf][bcol * LPAD + bkoff + 8]));
      acc[cb] = wmma_bf16(a, b, acc[cb]);
    }
    __syncthreads();
  }

  // epilogue: lane L, vgpr i -> row = wid*16 + 8*(L>>4) + i, col = (L&15)+16*cb
  const int rb  = m0 + wid * 16 + (lane >> 4) * 8;
  const int cb0 = n0 + (lane & 15);
#pragma unroll
  for (int cb = 0; cb < 4; ++cb) {
    int c = cb0 + cb * 16;
    float bv = (MODE != 0) ? bias[c & bmask] : 0.0f;
#pragma unroll
    for (int i = 0; i < 8; ++i) {
      float v = acc[cb][i] + bv;
      if (MODE == 1) v = fmaxf(v, 0.0f);
      C[(size_t)(rb + i) * Nc + c] = (bf16)v;
    }
  }
}

// ---------------------------------------------------------------------------
// Fused GRU: per 16x16 h-tile, compute 6 gate tiles (r/z/n x {ih,hh}) with
// WMMA (K = H = 128), then elementwise GRU update. Weights pre-transposed to
// (3H, H) so B fragments are contiguous K chunks. No LDS needed.
// Block = 256 thr = 8 waves; wave w handles column tile w (H=128 -> 8 tiles).
// ---------------------------------------------------------------------------
__global__ __launch_bounds__(256)
void gru_fused(const bf16* __restrict__ xg, const bf16* __restrict__ hbf,
               const float* __restrict__ hf,
               const bf16* __restrict__ WihT, const bf16* __restrict__ WhhT,
               const float* __restrict__ b_ih, const float* __restrict__ b_hh,
               float* __restrict__ hof, bf16* __restrict__ hobf) {
  const int H    = 128;
  const int t    = threadIdx.x;
  const int lane = t & 31;
  const int wid  = t >> 5;
  const int m0   = blockIdx.x * 16;

  const int arow  = m0 + (lane & 15);
  const int koff  = (lane >> 4) * 16;
  const int bkoff = (lane >> 4) * 16;
  const int bcol  = wid * 16 + (lane & 15);

  v8f acc[6] = {};
#pragma unroll
  for (int k0 = 0; k0 < H; k0 += 32) {
    v16bf ax = cat16(*(const v8bf*)(xg + (size_t)arow * H + k0 + koff),
                     *(const v8bf*)(xg + (size_t)arow * H + k0 + koff + 8));
    v16bf ah = cat16(*(const v8bf*)(hbf + (size_t)arow * H + k0 + koff),
                     *(const v8bf*)(hbf + (size_t)arow * H + k0 + koff + 8));
#pragma unroll
    for (int g = 0; g < 3; ++g) {
      const bf16* wi = WihT + (size_t)(g * H + bcol) * H + k0 + bkoff;
      acc[g] = wmma_bf16(ax, cat16(*(const v8bf*)wi, *(const v8bf*)(wi + 8)),
                         acc[g]);
      const bf16* wh = WhhT + (size_t)(g * H + bcol) * H + k0 + bkoff;
      acc[3 + g] = wmma_bf16(ah, cat16(*(const v8bf*)wh, *(const v8bf*)(wh + 8)),
                             acc[3 + g]);
    }
  }

  const int c = bcol;
  const float bir = b_ih[c],         bhr = b_hh[c];
  const float biz = b_ih[H + c],     bhz = b_hh[H + c];
  const float bin = b_ih[2 * H + c], bhn = b_hh[2 * H + c];
#pragma unroll
  for (int i = 0; i < 8; ++i) {
    int r = m0 + (lane >> 4) * 8 + i;
    float rg = sigmoidf(acc[0][i] + bir + acc[3][i] + bhr);
    float z  = sigmoidf(acc[1][i] + biz + acc[4][i] + bhz);
    float nn = tanhf(acc[2][i] + bin + rg * (acc[5][i] + bhn));
    float hn = (1.0f - z) * nn + z * hf[(size_t)r * H + c];
    hof[(size_t)r * H + c]  = hn;
    hobf[(size_t)r * H + c] = (bf16)hn;
  }
}

// --------------------------- small helper kernels ---------------------------
__global__ void f2bf_k(const float* __restrict__ s, bf16* __restrict__ d, int n) {
  int i = blockIdx.x * 256 + threadIdx.x;
  if (i < n) d[i] = (bf16)s[i];
}

// W (128 x 384) row-major -> WT (384 x 128) bf16 row-major
__global__ void transpose_w_k(const float* __restrict__ W, bf16* __restrict__ WT) {
  int i = blockIdx.x * 256 + threadIdx.x;  // < 49152
  int n = i >> 7, k = i & 127;
  WT[i] = (bf16)W[k * 384 + n];
}

// XW1[n, b*64+g] = x[b, s, n, 0] * W_g1[g]   (F == 1)
__global__ void xw1_x_k(const float* __restrict__ x, const float* __restrict__ Wg1,
                        bf16* __restrict__ XW1, int s) {
  int i = blockIdx.x * 256 + threadIdx.x;  // < 2048*2048
  int g = i & 63, b = (i >> 6) & 31, n = i >> 11;
  XW1[i] = (bf16)(x[(size_t)b * (12 * 2048) + s * 2048 + n] * Wg1[g]);
}

// XW1 from decoder output: outb layout r = n*32 + b
__global__ void xw1_o_k(const float* __restrict__ outb, const float* __restrict__ Wg1,
                        bf16* __restrict__ XW1) {
  int i = blockIdx.x * 256 + threadIdx.x;
  int g = i & 63, b = (i >> 6) & 31, n = i >> 11;
  XW1[i] = (bf16)(outb[n * 32 + b] * Wg1[g]);
}

// fc: out[b,n] = h[r]·W_fc + b_fc ; writes pred p to d_out[b, p, n]
__global__ void fc_k(const float* __restrict__ hf, const float* __restrict__ Wfc,
                     const float* __restrict__ bfc, float* __restrict__ outb,
                     float* __restrict__ dout, int p) {
  int r = blockIdx.x * 256 + threadIdx.x;  // < 65536, r = n*32 + b
  float a = bfc[0];
  const float* hr = hf + (size_t)r * 128;
#pragma unroll 8
  for (int k = 0; k < 128; ++k) a += hr[k] * Wfc[k];
  outb[r] = a;
  int b = r & 31, m = r >> 5;
  dout[(size_t)b * (3 * 2048) + p * 2048 + m] = a;
}

// ---------------------------------------------------------------------------
extern "C" void kernel_launch(void* const* d_in, const int* in_sizes, int n_in,
                              void* d_out, int out_size, void* d_ws, size_t ws_size,
                              hipStream_t stream) {
  (void)in_sizes; (void)n_in; (void)out_size; (void)ws_size;
  const float* x    = (const float*)d_in[0];
  const float* adj  = (const float*)d_in[1];
  const float* W_g1 = (const float*)d_in[2];
  const float* b_g1 = (const float*)d_in[3];
  const float* W_g2 = (const float*)d_in[4];
  const float* b_g2 = (const float*)d_in[5];
  const float* W_ih = (const float*)d_in[6];
  const float* W_hh = (const float*)d_in[7];
  const float* b_ih = (const float*)d_in[8];
  const float* b_hh = (const float*)d_in[9];
  const float* W_fc = (const float*)d_in[10];
  const float* b_fc = (const float*)d_in[11];
  float* out = (float*)d_out;

  const int N = 2048, B = 32, G = 64, H = 128;
  const size_t NB = (size_t)N * B;  // 65536

  char* p = (char*)d_ws;
  auto alloc = [&](size_t bytes) -> char* {
    char* r = p;
    p += (bytes + 255) & ~(size_t)255;
    return r;
  };
  bf16*  adj_bf  = (bf16*)alloc((size_t)N * N * 2);        //  8 MiB
  bf16*  XW1     = (bf16*)alloc((size_t)N * B * G * 2);    //  8 MiB
  bf16*  H1      = (bf16*)alloc((size_t)N * B * G * 2);    //  8 MiB
  bf16*  XG      = (bf16*)alloc((size_t)N * B * H * 2);    // 16 MiB
  bf16*  XGP     = (bf16*)alloc((size_t)N * B * H * 2);    // 16 MiB
  float* hf0     = (float*)alloc(NB * H * 4);              // 32 MiB
  float* hf1     = (float*)alloc(NB * H * 4);              // 32 MiB
  bf16*  hbf0    = (bf16*)alloc(NB * H * 2);               // 16 MiB
  bf16*  hbf1    = (bf16*)alloc(NB * H * 2);               // 16 MiB
  bf16*  WihT    = (bf16*)alloc((size_t)3 * H * H * 2);
  bf16*  WhhT    = (bf16*)alloc((size_t)3 * H * H * 2);
  bf16*  Wg2_bf  = (bf16*)alloc((size_t)G * H * 2);
  float* outb    = (float*)alloc(NB * 4);

  // one-time (per call) weight prep — deterministic
  f2bf_k<<<(N * N) / 256, 256, 0, stream>>>(adj, adj_bf, N * N);
  f2bf_k<<<(G * H + 255) / 256, 256, 0, stream>>>(W_g2, Wg2_bf, G * H);
  transpose_w_k<<<(3 * H * H) / 256, 256, 0, stream>>>(W_ih, WihT);
  transpose_w_k<<<(3 * H * H) / 256, 256, 0, stream>>>(W_hh, WhhT);
  hipMemsetAsync(hf0, 0, NB * H * 4, stream);
  hipMemsetAsync(hbf0, 0, NB * H * 2, stream);

  float* hf_cur = hf0;  float* hf_nxt = hf1;
  bf16* hbf_cur = hbf0; bf16* hbf_nxt = hbf1;

  auto cell = [&]() {
    // H1 = relu(adj @ XW1 + b_g1)            (2048 x 2048, K=2048)
    wmma_gemm_bf16<1><<<dim3((B * G) / BN, N / BM), 256, 0, stream>>>(
        adj_bf, XW1, H1, b_g1, N, B * G, N, G - 1);
    // XG = H1 @ W_g2                          (65536 x 128, K=64)
    wmma_gemm_bf16<0><<<dim3(H / BN, (int)(NB / BM)), 256, 0, stream>>>(
        H1, Wg2_bf, XG, nullptr, (int)NB, H, G, 0);
    // XGP = adj @ XG + b_g2                   (2048 x 4096, K=2048)
    wmma_gemm_bf16<2><<<dim3((B * H) / BN, N / BM), 256, 0, stream>>>(
        adj_bf, XG, XGP, b_g2, N, B * H, N, H - 1);
    // GRU update
    gru_fused<<<(int)(NB / 16), 256, 0, stream>>>(
        XGP, hbf_cur, hf_cur, WihT, WhhT, b_ih, b_hh, hf_nxt, hbf_nxt);
    { float* tf = hf_cur; hf_cur = hf_nxt; hf_nxt = tf; }
    { bf16* tb = hbf_cur; hbf_cur = hbf_nxt; hbf_nxt = tb; }
  };

  // encoder: 12 steps over x
  for (int s = 0; s < 12; ++s) {
    xw1_x_k<<<(N * B * G) / 256, 256, 0, stream>>>(x, W_g1, XW1, s);
    cell();
  }
  // decoder: 3 autoregressive steps (last cell result unused -> skipped)
  for (int ps = 0; ps < 3; ++ps) {
    fc_k<<<(int)(NB / 256), 256, 0, stream>>>(hf_cur, W_fc, b_fc, outb, out, ps);
    if (ps < 2) {
      xw1_o_k<<<(N * B * G) / 256, 256, 0, stream>>>(outb, W_g1, XW1);
      cell();
    }
  }
}
